// EuclideanDistance_36129264894473
// MI455X (gfx1250) — matrix operand order repaired and checked by reading, not compiled, not run on hardware
//
#include <hip/hip_runtime.h>

typedef __attribute__((ext_vector_type(8)))  __bf16 v8bf;
typedef __attribute__((ext_vector_type(16))) __bf16 v16bf;
typedef __attribute__((ext_vector_type(8)))  float  v8f;

#define K_DIM   64
#define M_BLK   128
#define N_BLK   64
#define THREADS 256
#define LDA     72   // padded LDS row stride in bf16 elems (144 B, 16B-aligned, bank-spread)

__device__ __forceinline__ v16bf load_frag16(const __bf16* p0, const __bf16* p1) {
    v8bf a = *(const v8bf*)p0;
    v8bf b = *(const v8bf*)p1;
    return __builtin_shufflevector(a, b, 0,1,2,3,4,5,6,7,8,9,10,11,12,13,14,15);
}

__global__ __launch_bounds__(THREADS)
void euclid_wmma_kernel(const float* __restrict__ mat1,
                        const float* __restrict__ mat2,
                        float* __restrict__ out,
                        int n1, int n2) {
    __shared__ __align__(16) __bf16 sAhi[M_BLK * LDA];
    __shared__ __align__(16) __bf16 sAlo[M_BLK * LDA];
    __shared__ __align__(16) __bf16 sBhi[N_BLK * LDA];
    __shared__ __align__(16) __bf16 sBlo[N_BLK * LDA];
    __shared__ float sSq1[M_BLK];
    __shared__ float sSq2[N_BLK];
    __shared__ float sRed[2 * THREADS];

    const int t  = threadIdx.x;
    const int m0 = blockIdx.y * M_BLK;   // row block of mat_1
    const int n0 = blockIdx.x * N_BLK;   // row block of mat_2 (columns of output)

    // ---------------- Phase 1: load + split-convert to bf16 hi/lo in LDS ----------------
    // A: 128 rows x 64 f32.  thread t -> row r = t/2, half h = t%2 (32 floats = 8 float4)
    {
        const int r  = t >> 1;
        const int h  = t & 1;
        const int gm = m0 + r;
        float sq = 0.0f;
        __bf16* dsthi = &sAhi[r * LDA + h * 32];
        __bf16* dstlo = &sAlo[r * LDA + h * 32];
        if (gm < n1) {
            const float4* p = (const float4*)(mat1 + (size_t)gm * K_DIM + h * 32);
            #pragma unroll
            for (int i = 0; i < 8; ++i) {
                float4 v = p[i];
                float vv[4] = {v.x, v.y, v.z, v.w};
                #pragma unroll
                for (int c = 0; c < 4; ++c) {
                    float a  = vv[c];
                    __bf16 hi = (__bf16)a;
                    __bf16 lo = (__bf16)(a - (float)hi);
                    dsthi[i * 4 + c] = hi;
                    dstlo[i * 4 + c] = lo;
                    sq += a * a;
                }
            }
        } else {
            #pragma unroll
            for (int i = 0; i < 32; ++i) { dsthi[i] = (__bf16)0.0f; dstlo[i] = (__bf16)0.0f; }
        }
        sRed[t] = sq;
    }
    // B: 64 rows x 64 f32.  thread t -> row rb = t/4, quarter q = t%4 (16 floats)
    {
        const int rb = t >> 2;
        const int q  = t & 3;
        const int gn = n0 + rb;
        float sq = 0.0f;
        __bf16* dsthi = &sBhi[rb * LDA + q * 16];
        __bf16* dstlo = &sBlo[rb * LDA + q * 16];
        if (gn < n2) {
            const float4* p = (const float4*)(mat2 + (size_t)gn * K_DIM + q * 16);
            #pragma unroll
            for (int i = 0; i < 4; ++i) {
                float4 v = p[i];
                float vv[4] = {v.x, v.y, v.z, v.w};
                #pragma unroll
                for (int c = 0; c < 4; ++c) {
                    float a  = vv[c];
                    __bf16 hi = (__bf16)a;
                    __bf16 lo = (__bf16)(a - (float)hi);
                    dsthi[i * 4 + c] = hi;
                    dstlo[i * 4 + c] = lo;
                    sq += a * a;
                }
            }
        } else {
            #pragma unroll
            for (int i = 0; i < 16; ++i) { dsthi[i] = (__bf16)0.0f; dstlo[i] = (__bf16)0.0f; }
        }
        sRed[THREADS + t] = sq;
    }
    __syncthreads();
    if (t < M_BLK) {
        sSq1[t] = sRed[2 * t] + sRed[2 * t + 1];
    } else if (t < M_BLK + N_BLK) {
        int n = t - M_BLK;
        sSq2[n] = sRed[THREADS + 4 * n + 0] + sRed[THREADS + 4 * n + 1] +
                  sRed[THREADS + 4 * n + 2] + sRed[THREADS + 4 * n + 3];
    }
    __syncthreads();

    // ---------------- Phase 2: WMMA bf16x3 GEMM ----------------
    const int w    = t >> 5;           // wave 0..7 (wave32)
    const int lane = t & 31;
    const int lh   = lane & 15;
    const int h16  = lane >> 4;        // 0: lanes 0-15, 1: lanes 16-31
    const int wr   = w >> 1;           // 0..3  -> M tiles {wr, wr+4}
    const int wc   = w & 1;            // 0..1  -> N tiles {wc, wc+2}
    const int mt[2] = {wr, wr + 4};
    const int nt[2] = {wc, wc + 2};

    v8f acc[2][2] = {};

    #pragma unroll
    for (int kc = 0; kc < 2; ++kc) {
        // A fragments (16x32 bf16): elems 0-7 = K kc*32 + h16*8 .. +7 ; elems 8-15 = +16
        v16bf ahi[2], alo[2], bhi[2], blo[2];
        #pragma unroll
        for (int i = 0; i < 2; ++i) {
            const int row  = mt[i] * 16 + lh;
            const int koff = kc * 32 + h16 * 8;
            ahi[i] = load_frag16(&sAhi[row * LDA + koff], &sAhi[row * LDA + koff + 16]);
            alo[i] = load_frag16(&sAlo[row * LDA + koff], &sAlo[row * LDA + koff + 16]);
        }
        // B fragments (32x16 bf16): elems 0-7 = K kc*32 + h16*8 .. +7 ; elems 8-15 = +16
        #pragma unroll
        for (int j = 0; j < 2; ++j) {
            const int row  = nt[j] * 16 + lh;
            const int koff = kc * 32 + h16 * 8;
            bhi[j] = load_frag16(&sBhi[row * LDA + koff], &sBhi[row * LDA + koff + 16]);
            blo[j] = load_frag16(&sBlo[row * LDA + koff], &sBlo[row * LDA + koff + 16]);
        }
        #pragma unroll
        for (int i = 0; i < 2; ++i) {
            #pragma unroll
            for (int j = 0; j < 2; ++j) {
                acc[i][j] = __builtin_amdgcn_wmma_f32_16x16x32_bf16(
                    false, ahi[i], false, bhi[j], (short)0, acc[i][j], false, false);
                acc[i][j] = __builtin_amdgcn_wmma_f32_16x16x32_bf16(
                    false, ahi[i], false, blo[j], (short)0, acc[i][j], false, false);
                acc[i][j] = __builtin_amdgcn_wmma_f32_16x16x32_bf16(
                    false, alo[i], false, bhi[j], (short)0, acc[i][j], false, false);
            }
        }
    }

    // ---------------- Epilogue: dist = sq1 + sq2 - 2*cross, non-temporal stream out ----------------
    #pragma unroll
    for (int i = 0; i < 2; ++i) {
        const int mbase = mt[i] * 16 + h16 * 8;     // local M for VGPR v: mbase + v
        #pragma unroll
        for (int j = 0; j < 2; ++j) {
            const int nl = nt[j] * 16 + lh;
            const int gn = n0 + nl;
            if (gn < n2) {
                const float s2 = sSq2[nl];
                #pragma unroll
                for (int v = 0; v < 8; ++v) {
                    const int gm = m0 + mbase + v;
                    if (gm < n1) {
                        float d = sSq1[mbase + v] + s2 - 2.0f * acc[i][j][v];
                        __builtin_nontemporal_store(d, out + (size_t)gm * n2 + gn);
                    }
                }
            }
        }
    }
}

extern "C" void kernel_launch(void* const* d_in, const int* in_sizes, int n_in,
                              void* d_out, int out_size, void* d_ws, size_t ws_size,
                              hipStream_t stream) {
    const float* mat1 = (const float*)d_in[0];
    const float* mat2 = (const float*)d_in[1];
    float* out = (float*)d_out;
    const int n1 = in_sizes[0] / K_DIM;   // 100000
    const int n2 = in_sizes[1] / K_DIM;   // 2048
    dim3 grid((n2 + N_BLK - 1) / N_BLK,   // x = N chunks (32): sweeps N fastest, A tile stays hot
              (n1 + M_BLK - 1) / M_BLK);  // y = M chunks (782)
    euclid_wmma_kernel<<<grid, THREADS, 0, stream>>>(mat1, mat2, out, n1, n2);
}
